// BinarizedMLP_11355893531204
// MI455X (gfx1250) — matrix operand order repaired and checked by reading, not compiled, not run on hardware
//
#include <hip/hip_runtime.h>
#include <hip/hip_bf16.h>

// Binarized MLP: y = relu(x @ sign(W)^T + b) x3, then final linear.
// f16 WMMA (weights exact +/-1 in f16, f32 accumulation). LDS double-buffered
// GEMM: block tile 128x128 (8 waves, each 32x64 = 2x4 wmma fragments),
// K-step 32 staged through LDS (padded rows -> conflict-free fragment reads).
// Main K-loop has unconditional prefetch/store/barrier; last step is peeled.

typedef __attribute__((ext_vector_type(16))) _Float16 v16h;
typedef __attribute__((ext_vector_type(8)))  _Float16 v8h;
typedef __attribute__((ext_vector_type(8)))  float    v8f;

// ---------------------------------------------------------------------------
// Prep: convert / binarize fp32 -> f16 with zero padding to [Rp x Cp].
// ---------------------------------------------------------------------------
__global__ __launch_bounds__(256)
void pad_convert_kernel(const float* __restrict__ src, _Float16* __restrict__ dst,
                        int R, int C, int Rp, int Cp, int binarize)
{
    int idx = blockIdx.x * blockDim.x + threadIdx.x;
    int total = Rp * Cp;
    if (idx >= total) return;
    int r = idx / Cp;
    int c = idx - r * Cp;
    float v = 0.0f;
    if (r < R && c < C) {
        v = src[(size_t)r * C + c];
        if (binarize) v = (v > 0.0f) ? 1.0f : ((v < 0.0f) ? -1.0f : 0.0f);
    }
    dst[idx] = (_Float16)v;
}

// ---------------------------------------------------------------------------
// WMMA GEMM with LDS double buffering.
//   out[M x ldo] = A[M x K] * B[Npad x K]^T + bias  (optional ReLU)
// A, B row-major f16, K multiple of 32, Npad multiple of 128.
// Block: 256 threads = 8 waves (wm 0..3, wn 0..1); wave tile 32(M) x 64(N).
// ---------------------------------------------------------------------------
template<bool RELU, bool HALF_OUT>
__global__ __launch_bounds__(256)
void bin_gemm_wmma(const _Float16* __restrict__ A, int lda,
                   const _Float16* __restrict__ B, int ldb,
                   const float* __restrict__ bias, int nbias,
                   _Float16* __restrict__ outH, float* __restrict__ outF,
                   int ldo, int nstore, int K)
{
    constexpr int SAW = 40;  // LDS row stride (halves): 32 data + 8 pad (80 B)
    __shared__ _Float16 As[2][128 * SAW];
    __shared__ _Float16 Bs[2][128 * SAW];

    const int t    = threadIdx.x;
    const int lane = t & 31;             // wave32
    const int wave = t >> 5;
    const int wm   = wave >> 1;          // 0..3
    const int wn   = wave & 1;           // 0..1
    const int lr   = lane & 15;
    const bool hi  = (lane >= 16);
    const int kaoff = hi ? 8  : 0;       // A frag: K chunks {0..7,16..23}/{8..15,24..31}
    const int kboff = hi ? 8  : 0;       // B frag: K chunks {0..7,16..23}/{8..15,24..31}

    // Staging: 128x32 tile = 512 x 16B chunks; each thread moves chunks t, t+256.
    const int srow = t >> 2;             // 0..63
    const int scol = (t & 3) * 8;        // 0,8,16,24 (halves)

    const _Float16* Ag0 = A + (size_t)(blockIdx.x * 128 + srow) * lda + scol;
    const _Float16* Ag1 = Ag0 + (size_t)64 * lda;
    const _Float16* Bg0 = B + (size_t)(blockIdx.y * 128 + srow) * ldb + scol;
    const _Float16* Bg1 = Bg0 + (size_t)64 * ldb;

    const int s0 = srow * SAW + scol;
    const int s1 = (srow + 64) * SAW + scol;

    v8f acc[2][4] = {};

    // Fragment-load + 8x WMMA from LDS buffer `buf`.
    auto do_tile = [&](int buf) {
        union frag { v16h v; v8h h[2]; };
        frag a[2], b[4];
        #pragma unroll
        for (int tm = 0; tm < 2; ++tm) {
            const _Float16* p = &As[buf][(wm * 32 + tm * 16 + lr) * SAW];
            a[tm].h[0] = *(const v8h*)(p + kaoff);
            a[tm].h[1] = *(const v8h*)(p + 16 + kaoff);
        }
        #pragma unroll
        for (int tn = 0; tn < 4; ++tn) {
            const _Float16* p = &Bs[buf][(wn * 64 + tn * 16 + lr) * SAW + kboff];
            b[tn].h[0] = *(const v8h*)(p);
            b[tn].h[1] = *(const v8h*)(p + 16);
        }
        #pragma unroll
        for (int tm = 0; tm < 2; ++tm)
            #pragma unroll
            for (int tn = 0; tn < 4; ++tn)
                acc[tm][tn] = __builtin_amdgcn_wmma_f32_16x16x32_f16(
                    false, a[tm].v, false, b[tn].v, (short)0, acc[tm][tn],
                    false, false);
    };

    // Prologue: stage tile 0 into buffer 0.
    {
        v8h pa0 = *(const v8h*)(Ag0);
        v8h pa1 = *(const v8h*)(Ag1);
        v8h pb0 = *(const v8h*)(Bg0);
        v8h pb1 = *(const v8h*)(Bg1);
        *(v8h*)&As[0][s0] = pa0;
        *(v8h*)&As[0][s1] = pa1;
        *(v8h*)&Bs[0][s0] = pb0;
        *(v8h*)&Bs[0][s1] = pb1;
    }
    __syncthreads();

    const int S = K >> 5;
    int s = 0;
    // Main loop: unconditional prefetch -> compute -> store other buffer.
    for (; s + 1 < S; ++s) {
        const int kk = (s + 1) << 5;
        v8h pa0 = *(const v8h*)(Ag0 + kk);
        v8h pa1 = *(const v8h*)(Ag1 + kk);
        v8h pb0 = *(const v8h*)(Bg0 + kk);
        v8h pb1 = *(const v8h*)(Bg1 + kk);

        do_tile(s & 1);

        const int dbuf = (s & 1) ^ 1;
        *(v8h*)&As[dbuf][s0] = pa0;
        *(v8h*)&As[dbuf][s1] = pa1;
        *(v8h*)&Bs[dbuf][s0] = pb0;
        *(v8h*)&Bs[dbuf][s1] = pb1;
        __syncthreads();
    }
    // Peeled last step: compute only.
    do_tile(s & 1);

    // Epilogue: C layout — VGPR r: lanes 0-15 M=r, lanes 16-31 M=8+r; N=lane&15.
    const int r0 = blockIdx.x * 128 + wm * 32;
    const int c0 = blockIdx.y * 128 + wn * 64;
    const int mrow = hi ? 8 : 0;
    #pragma unroll
    for (int tm = 0; tm < 2; ++tm) {
        #pragma unroll
        for (int tn = 0; tn < 4; ++tn) {
            int ng = c0 + tn * 16 + lr;
            int mg = r0 + tm * 16 + mrow;
            float bv = (ng < nbias) ? bias[ng] : 0.0f;
            #pragma unroll
            for (int r = 0; r < 8; ++r) {
                float v = acc[tm][tn][r] + bv;
                if (RELU) v = fmaxf(v, 0.0f);
                if (ng < nstore) {
                    size_t o = (size_t)(mg + r) * ldo + ng;
                    if (HALF_OUT) outH[o] = (_Float16)v;
                    else          outF[o] = v;
                }
            }
        }
    }
}

// ---------------------------------------------------------------------------
// Launch: prep (x->f16 pad, sign(W)->f16 pad) then 4 fused GEMM layers.
// DIMS = [784, 1024, 512, 256, 10], M = 16384. Kpad(784)=800, W4 Npad=128.
// ---------------------------------------------------------------------------
extern "C" void kernel_launch(void* const* d_in, const int* in_sizes, int n_in,
                              void* d_out, int out_size, void* d_ws, size_t ws_size,
                              hipStream_t stream)
{
    (void)in_sizes; (void)n_in; (void)out_size; (void)ws_size;

    const float* x  = (const float*)d_in[0];
    const float* w1 = (const float*)d_in[1];
    const float* b1 = (const float*)d_in[2];
    const float* w2 = (const float*)d_in[3];
    const float* b2 = (const float*)d_in[4];
    const float* w3 = (const float*)d_in[5];
    const float* b3 = (const float*)d_in[6];
    const float* w4 = (const float*)d_in[7];
    const float* b4 = (const float*)d_in[8];
    float* out = (float*)d_out;

    const int M = 16384;

    // Workspace carve-up (256B aligned; ~88 MB total).
    char* ws = (char*)d_ws;
    size_t off = 0;
    auto carve = [&](size_t bytes) -> void* {
        void* p = ws + off;
        off += (bytes + 255) & ~(size_t)255;
        return p;
    };
    _Float16* xh  = (_Float16*)carve((size_t)M * 800 * 2);
    _Float16* w1h = (_Float16*)carve((size_t)1024 * 800 * 2);
    _Float16* w2h = (_Float16*)carve((size_t)512 * 1024 * 2);
    _Float16* w3h = (_Float16*)carve((size_t)256 * 512 * 2);
    _Float16* w4h = (_Float16*)carve((size_t)128 * 256 * 2);
    _Float16* a1  = (_Float16*)carve((size_t)M * 1024 * 2);
    _Float16* a2  = (_Float16*)carve((size_t)M * 512 * 2);
    _Float16* a3  = (_Float16*)carve((size_t)M * 256 * 2);

    dim3 blk(256);
    auto grid1 = [](size_t n) { return dim3((unsigned)((n + 255) / 256)); };

    // Prep: pad/convert activations and binarize weights.
    pad_convert_kernel<<<grid1((size_t)M * 800), blk, 0, stream>>>(x,  xh,  M,    784,  M,    800,  0);
    pad_convert_kernel<<<grid1((size_t)1024 * 800), blk, 0, stream>>>(w1, w1h, 1024, 784,  1024, 800,  1);
    pad_convert_kernel<<<grid1((size_t)512 * 1024), blk, 0, stream>>>(w2, w2h, 512,  1024, 512,  1024, 1);
    pad_convert_kernel<<<grid1((size_t)256 * 512), blk, 0, stream>>>(w3, w3h, 256,  512,  256,  512,  1);
    pad_convert_kernel<<<grid1((size_t)128 * 256), blk, 0, stream>>>(w4, w4h, 10,   256,  128,  256,  1);

    // Layer 1: [16384 x 800] * [1024 x 800]^T -> a1 (relu, f16)
    bin_gemm_wmma<true, true><<<dim3(M / 128, 1024 / 128), blk, 0, stream>>>(
        xh, 800, w1h, 800, b1, 1024, a1, nullptr, 1024, 1024, 800);
    // Layer 2: [16384 x 1024] * [512 x 1024]^T -> a2 (relu, f16)
    bin_gemm_wmma<true, true><<<dim3(M / 128, 512 / 128), blk, 0, stream>>>(
        a1, 1024, w2h, 1024, b2, 512, a2, nullptr, 512, 512, 1024);
    // Layer 3: [16384 x 512] * [256 x 512]^T -> a3 (relu, f16)
    bin_gemm_wmma<true, true><<<dim3(M / 128, 256 / 128), blk, 0, stream>>>(
        a2, 512, w3h, 512, b3, 256, a3, nullptr, 256, 256, 512);
    // Layer 4: [16384 x 256] * [128 x 256]^T -> out fp32, only n < 10 stored
    bin_gemm_wmma<false, false><<<dim3(M / 128, 1), blk, 0, stream>>>(
        a3, 256, w4h, 256, b4, 10, nullptr, out, 10, 10, 256);
}
